// GPT2Model_34376918238084
// MI455X (gfx1250) — compile-verified, matrix operand order checked by reading, not run on hardware
//
#include <hip/hip_runtime.h>
#include <hip/hip_bf16.h>
#include <math.h>

// ---------------- model dims ----------------
constexpr int VOCAB  = 50257;
constexpr int VPAD   = 50304;  // VOCAB padded to multiple of 128
constexpr int DM     = 1024;
constexpr int NH     = 16;
constexpr int NL     = 6;
constexpr int DFF    = 4096;
constexpr int SEQ    = 1024;
constexpr int BATCH  = 2;
constexpr int DK     = 64;
constexpr int MTOK   = BATCH * SEQ;   // 2048 rows

#ifndef USE_ASYNC_LDS
#define USE_ASYNC_LDS 1
#endif

typedef unsigned short u16;
typedef __attribute__((ext_vector_type(16))) __bf16 v16bf;
typedef __attribute__((ext_vector_type(8)))  float  v8f;

union FragU { unsigned int u[8]; v16bf v; };

__device__ __forceinline__ u16 f2bf(float f) {
    unsigned int u = __float_as_uint(f);
    u += 0x7FFFu + ((u >> 16) & 1u);        // round-to-nearest-even
    return (u16)(u >> 16);
}
__device__ __forceinline__ unsigned int pk2(float a, float b) {
    return (unsigned int)f2bf(a) | ((unsigned int)f2bf(b) << 16);
}

// 16B global -> LDS copy. Async path uses the CDNA5 ASYNCcnt data mover
// (global_load_async_to_lds_b128, CDNA5 ISA).
__device__ __forceinline__ void cp16(void* ldsDst, const void* gBase, unsigned byteOff) {
#if USE_ASYNC_LDS
    unsigned laddr = (unsigned)(unsigned long long)ldsDst;   // low 32b = LDS offset
    asm volatile("global_load_async_to_lds_b128 %0, %1, %2 offset:0"
                 :: "v"(laddr), "v"(byteOff), "s"(gBase)
                 : "memory");
#else
    *reinterpret_cast<uint4*>(ldsDst) =
        *reinterpret_cast<const uint4*>((const char*)gBase + byteOff);
#endif
}
__device__ __forceinline__ void cp16_wait() {
#if USE_ASYNC_LDS
    asm volatile("s_wait_asynccnt 0x0" ::: "memory");
#endif
}

// Fragment gather from an LDS row of contiguous bf16. Pair p -> K offset
// Kp = (p<4) ? kh+2p : 16+kh+2(p-4)   (ISA 7.12.2 16-bit A/B layout).
__device__ __forceinline__ v16bf frag_ld(const u16* base, int kh) {
    FragU u;
#pragma unroll
    for (int p = 0; p < 8; ++p) {
        int Kp = (p < 4) ? (kh + 2 * p) : (16 + kh + 2 * (p - 4));
        u.u[p] = *reinterpret_cast<const unsigned int*>(base + Kp);
    }
    return u.v;
}

// ================= embedding + positional encoding =================
__global__ void embed_kernel(const int* __restrict__ ids,
                             const float* __restrict__ emb,
                             float* __restrict__ x) {
    int tok = blockIdx.x;
    int s   = tok % SEQ;
    int id  = ids[tok];
    int d0  = threadIdx.x * 4;
    const float c0 = 9.210340372f / (float)DM;   // ln(10000)/D
#pragma unroll
    for (int c = 0; c < 4; ++c) {
        int d = d0 + c;
        float freq = __expf(-(float)(d & ~1) * c0);
        float ang  = (float)s * freq;
        float pe   = (d & 1) ? __cosf(ang) : __sinf(ang);
        x[(size_t)tok * DM + d] = emb[(size_t)id * DM + d] * 32.0f + pe; // sqrt(1024)
    }
}

// ================= fp32 -> bf16 bulk convert =================
__global__ void cvt_kernel(const float* __restrict__ src, u16* __restrict__ dst, long n) {
    long i0 = ((long)blockIdx.x * 256 + threadIdx.x) * 8;
    if (i0 + 8 <= n) {
        float4 a = *reinterpret_cast<const float4*>(src + i0);
        float4 b = *reinterpret_cast<const float4*>(src + i0 + 4);
        unsigned int* d = reinterpret_cast<unsigned int*>(dst + i0);
        d[0] = pk2(a.x, a.y); d[1] = pk2(a.z, a.w);
        d[2] = pk2(b.x, b.y); d[3] = pk2(b.z, b.w);
    } else {
        for (int k = 0; k < 8; ++k)
            if (i0 + k < n) dst[i0 + k] = f2bf(src[i0 + k]);
    }
}

// ========= fp32 [K x N] -> bf16 [N x K] convert + transpose (64x64 LDS tile) =========
__global__ __launch_bounds__(256) void cvtT_kernel(const float* __restrict__ src,
                                                   u16* __restrict__ dst, int K, int N) {
    __shared__ float t[64][65];
    const int tid = threadIdx.x;
    const int nBase = blockIdx.x * 64;
    const int kBase = blockIdx.y * 64;
#pragma unroll
    for (int j = 0; j < 4; ++j) {
        int f = tid * 4 + j;            // 1024 float4s = 64 rows x 16
        int r = f >> 4;
        int c4 = (f & 15) << 2;
        float4 q = *reinterpret_cast<const float4*>(src + (size_t)(kBase + r) * N + nBase + c4);
        t[r][c4 + 0] = q.x; t[r][c4 + 1] = q.y; t[r][c4 + 2] = q.z; t[r][c4 + 3] = q.w;
    }
    __syncthreads();
#pragma unroll
    for (int j = 0; j < 8; ++j) {
        int u = tid * 8 + j;            // 2048 bf16-pairs
        int n = u >> 5;
        int kk = (u & 31) << 1;
        unsigned int val = pk2(t[kk][n], t[kk + 1][n]);
        *reinterpret_cast<unsigned int*>(dst + (size_t)(nBase + n) * K + kBase + kk) = val;
    }
}

// ================= residual + layernorm (fp32) =================
template <bool HAS_RES>
__global__ void addln_kernel(float* __restrict__ x, const float* __restrict__ res,
                             const float* __restrict__ w, const float* __restrict__ b) {
    __shared__ float red[8];
    int row = blockIdx.x, tid = threadIdx.x;
    size_t base = (size_t)row * DM + tid * 4;
    float4 xv = *reinterpret_cast<float4*>(x + base);
    if (HAS_RES) {
        float4 rv = *reinterpret_cast<const float4*>(res + base);
        xv.x += rv.x; xv.y += rv.y; xv.z += rv.z; xv.w += rv.w;
    }
    float s = xv.x + xv.y + xv.z + xv.w;
#pragma unroll
    for (int off = 16; off; off >>= 1) s += __shfl_xor(s, off, 32);
    if ((tid & 31) == 0) red[tid >> 5] = s;
    __syncthreads();
    float tot = 0.f;
#pragma unroll
    for (int i = 0; i < 8; ++i) tot += red[i];
    float mu = tot * (1.0f / (float)DM);
    __syncthreads();
    float dx0 = xv.x - mu, dx1 = xv.y - mu, dx2 = xv.z - mu, dx3 = xv.w - mu;
    s = dx0 * dx0 + dx1 * dx1 + dx2 * dx2 + dx3 * dx3;
#pragma unroll
    for (int off = 16; off; off >>= 1) s += __shfl_xor(s, off, 32);
    if ((tid & 31) == 0) red[tid >> 5] = s;
    __syncthreads();
    tot = 0.f;
#pragma unroll
    for (int i = 0; i < 8; ++i) tot += red[i];
    float inv = rsqrtf(tot * (1.0f / (float)DM) + 1e-5f);
    float4 wv = *reinterpret_cast<const float4*>(w + tid * 4);
    float4 bv = *reinterpret_cast<const float4*>(b + tid * 4);
    float4 o;
    o.x = dx0 * inv * wv.x + bv.x;
    o.y = dx1 * inv * wv.y + bv.y;
    o.z = dx2 * inv * wv.z + bv.z;
    o.w = dx3 * inv * wv.w + bv.w;
    *reinterpret_cast<float4*>(x + base) = o;
}

// ================= WMMA bf16 GEMM =================
// A: bf16 [M x K] row-major.  B: bf16 [N x K] row-major (pre-transposed weights).
// EPI: 0 none, 1 +bias, 2 +bias+GELU.  OUTBF: store bf16 instead of fp32.
// Block 256 thr = 8 waves, tile 128x128x64 (2 WMMA K-steps per stage, 16 wmma/barrier).
template <int EPI, bool OUTBF>
__global__ __launch_bounds__(256) void gemm_kernel(const u16* __restrict__ A,
                                                   const u16* __restrict__ Bm,
                                                   const float* __restrict__ bias,
                                                   void* __restrict__ Cv,
                                                   int N, int K) {
    __shared__ u16 lA[128 * 64];      // [row][k]
    __shared__ u16 lB[128 * 64];      // [n][k]
    const int tid = threadIdx.x;
    const int lane = tid & 31;
    const int wid = tid >> 5;
    const int wm = wid & 3;           // 4 waves along M (32 rows)
    const int wn = wid >> 2;          // 2 waves along N (64 cols)
    const int mBase = blockIdx.y * 128;
    const int nBase = blockIdx.x * 128;
    const int g  = lane >> 4;
    const int kh = g * 8;
    const int ln = lane & 15;

    v8f acc[2][4];
#pragma unroll
    for (int mi = 0; mi < 2; ++mi)
#pragma unroll
        for (int ni = 0; ni < 4; ++ni)
            acc[mi][ni] = (v8f){0.f, 0.f, 0.f, 0.f, 0.f, 0.f, 0.f, 0.f};

    for (int kt = 0; kt < K; kt += 64) {
        // stage A and B 128x64 bf16 tiles: 1024 16B-chunks each, 4 per thread
#pragma unroll
        for (int j = 0; j < 4; ++j) {
            int c = tid * 4 + j;
            int row = c >> 3;             // 8 chunks per 64-elem row
            int q = (c & 7) << 3;
            cp16(&lA[row * 64 + q], A, (unsigned)(((size_t)(mBase + row) * K + kt + q) * 2));
            cp16(&lB[row * 64 + q], Bm, (unsigned)(((size_t)(nBase + row) * K + kt + q) * 2));
        }
        cp16_wait();
        __syncthreads();

#pragma unroll
        for (int ks = 0; ks < 64; ks += 32) {
            v16bf af[2], bf[4];
#pragma unroll
            for (int mi = 0; mi < 2; ++mi)
                af[mi] = frag_ld(&lA[(wm * 32 + mi * 16 + ln) * 64 + ks], kh);
#pragma unroll
            for (int ni = 0; ni < 4; ++ni)
                bf[ni] = frag_ld(&lB[(wn * 64 + ni * 16 + ln) * 64 + ks], kh);
#pragma unroll
            for (int mi = 0; mi < 2; ++mi)
#pragma unroll
                for (int ni = 0; ni < 4; ++ni)
                    acc[mi][ni] = __builtin_amdgcn_wmma_f32_16x16x32_bf16(
                        false, af[mi], false, bf[ni], (short)0, acc[mi][ni], false, false);
        }
        __syncthreads();
    }

    // epilogue
#pragma unroll
    for (int ni = 0; ni < 4; ++ni) {
        int col = nBase + wn * 64 + ni * 16 + ln;
        float bv = 0.f;
        if (EPI >= 1) bv = bias[col];
#pragma unroll
        for (int mi = 0; mi < 2; ++mi) {
            int rowb = mBase + wm * 32 + mi * 16 + g * 8;
#pragma unroll
            for (int r = 0; r < 8; ++r) {
                float v = acc[mi][ni][r] + bv;
                if (EPI == 2) v = 0.5f * v * (1.0f + erff(v * 0.70710678f));
                if (col < N) {
                    if (OUTBF)
                        ((u16*)Cv)[(size_t)(rowb + r) * N + col] = f2bf(v);
                    else
                        ((float*)Cv)[(size_t)(rowb + r) * N + col] = v;
                }
            }
        }
    }
}

// ================= fused causal attention (flash-style, bf16 in/out) =================
// Grid: B*H*(S/64), 128 threads (4 waves), wave = 16 query rows.
__global__ __launch_bounds__(128) void attn_kernel(const u16* __restrict__ qp,
                                                   const u16* __restrict__ kp,
                                                   const u16* __restrict__ vp,
                                                   u16* __restrict__ outp) {
    __shared__ u16 lKt[32 * 64];        // [key][feat]
    __shared__ u16 lVt[64 * 32];        // [feat][key] (transposed)
    __shared__ u16 lP[4][16][32];       // per-wave P scratch

    const int qblocks = SEQ / 64;
    int bh = blockIdx.x / qblocks;
    int qb = (blockIdx.x % qblocks) * 64;
    int b = bh / NH, h = bh % NH;
    const int tid = threadIdx.x;
    const int lane = tid & 31;
    const int wid = tid >> 5;
    const int g  = lane >> 4;
    const int kh = g * 8;
    const int ln = lane & 15;
    const size_t bS = (size_t)b * SEQ;
    const int hoff = h * DK;

    // Q fragments straight from global bf16 (pairs contiguous along features)
    v16bf qf[2];
    {
        int qrow = qb + wid * 16 + ln;
        const u16* qr = qp + (bS + qrow) * DM + hoff;
#pragma unroll
        for (int c = 0; c < 2; ++c) {
            FragU u;
#pragma unroll
            for (int p = 0; p < 8; ++p) {
                int Kp = (p < 4) ? (kh + 2 * p) : (16 + kh + 2 * (p - 4));
                u.u[p] = *reinterpret_cast<const unsigned int*>(qr + c * 32 + Kp);
            }
            qf[c] = u.v;
        }
    }

    v8f oacc[4];
#pragma unroll
    for (int nt = 0; nt < 4; ++nt)
        oacc[nt] = (v8f){0.f, 0.f, 0.f, 0.f, 0.f, 0.f, 0.f, 0.f};
    float mrow[8], lrow[8];
#pragma unroll
    for (int r = 0; r < 8; ++r) { mrow[r] = -1e30f; lrow[r] = 0.f; }

    const int kbEnd = qb + 64;
    for (int kb = 0; kb < kbEnd; kb += 32) {
        // K tile: pure bf16 copy (async path); V tile: transpose via VGPRs
#pragma unroll
        for (int j = 0; j < 2; ++j) {
            int c = tid * 2 + j;          // 256 chunks
            int key = c >> 3;
            int q = (c & 7) << 3;
            size_t ge = (bS + kb + key) * (size_t)DM + hoff + q;
            cp16(&lKt[key * 64 + q], kp, (unsigned)(ge * 2));
            uint4 w = *reinterpret_cast<const uint4*>(vp + ge);
            lVt[(q + 0) * 32 + key] = (u16)(w.x);
            lVt[(q + 1) * 32 + key] = (u16)(w.x >> 16);
            lVt[(q + 2) * 32 + key] = (u16)(w.y);
            lVt[(q + 3) * 32 + key] = (u16)(w.y >> 16);
            lVt[(q + 4) * 32 + key] = (u16)(w.z);
            lVt[(q + 5) * 32 + key] = (u16)(w.z >> 16);
            lVt[(q + 6) * 32 + key] = (u16)(w.w);
            lVt[(q + 7) * 32 + key] = (u16)(w.w >> 16);
        }
        cp16_wait();
        __syncthreads();

        // scores 16x32 (two C-frags), K-dim 64 in 2 chunks
        v8f z = {0.f, 0.f, 0.f, 0.f, 0.f, 0.f, 0.f, 0.f};
        v8f sc0 = z, sc1 = z;
#pragma unroll
        for (int c = 0; c < 2; ++c) {
            v16bf b0 = frag_ld(&lKt[(0 * 16 + ln) * 64 + c * 32], kh);
            v16bf b1 = frag_ld(&lKt[(1 * 16 + ln) * 64 + c * 32], kh);
            sc0 = __builtin_amdgcn_wmma_f32_16x16x32_bf16(false, qf[c], false, b0,
                                                          (short)0, sc0, false, false);
            sc1 = __builtin_amdgcn_wmma_f32_16x16x32_bf16(false, qf[c], false, b1,
                                                          (short)0, sc1, false, false);
        }

        // causal mask + online softmax (1/sqrt(64) applied post-WMMA in fp32)
#pragma unroll
        for (int r = 0; r < 8; ++r) {
            int qrow = qb + wid * 16 + r + 8 * g;
            float s0 = sc0[r] * 0.125f; if (kb + ln > qrow)      s0 = -1e30f;
            float s1 = sc1[r] * 0.125f; if (kb + 16 + ln > qrow) s1 = -1e30f;
            float rv = fmaxf(s0, s1);
#pragma unroll
            for (int off = 8; off; off >>= 1) rv = fmaxf(rv, __shfl_xor(rv, off, 16));
            float mnew = fmaxf(mrow[r], rv);
            float scale = __expf(mrow[r] - mnew);
            float p0 = __expf(s0 - mnew);
            float p1 = __expf(s1 - mnew);
            float rs = p0 + p1;
#pragma unroll
            for (int off = 8; off; off >>= 1) rs += __shfl_xor(rs, off, 16);
            lrow[r] = lrow[r] * scale + rs;
            mrow[r] = mnew;
#pragma unroll
            for (int nt = 0; nt < 4; ++nt) oacc[nt][r] *= scale;
            lP[wid][r + 8 * g][ln]      = f2bf(p0);
            lP[wid][r + 8 * g][16 + ln] = f2bf(p1);
        }

        // O += P (16x32) x V (32x64); wave-local LDS re-layout of P
        v16bf pa = frag_ld(&lP[wid][ln][0], kh);
#pragma unroll
        for (int nt = 0; nt < 4; ++nt) {
            v16bf bv = frag_ld(&lVt[(nt * 16 + ln) * 32], kh);
            oacc[nt] = __builtin_amdgcn_wmma_f32_16x16x32_bf16(false, pa, false, bv,
                                                               (short)0, oacc[nt], false, false);
        }
        __syncthreads();
    }

    // normalize and store bf16
#pragma unroll
    for (int nt = 0; nt < 4; ++nt)
#pragma unroll
        for (int r = 0; r < 8; ++r) {
            int qrow = qb + wid * 16 + r + 8 * g;
            int col = hoff + nt * 16 + ln;
            outp[(bS + qrow) * DM + col] = f2bf(oacc[nt][r] / lrow[r]);
        }
}

// ================= host-side orchestration =================
extern "C" void kernel_launch(void* const* d_in, const int* in_sizes, int n_in,
                              void* d_out, int out_size, void* d_ws, size_t ws_size,
                              hipStream_t stream) {
    (void)in_sizes; (void)n_in; (void)out_size; (void)ws_size;
    const int*   ids = (const int*)d_in[0];
    const float* emb = (const float*)d_in[1];
    const float* wq  = (const float*)d_in[2];
    const float* wk  = (const float*)d_in[3];
    const float* wv  = (const float*)d_in[4];
    const float* wo  = (const float*)d_in[5];
    const float* w1  = (const float*)d_in[6];
    const float* b1  = (const float*)d_in[7];
    const float* w2  = (const float*)d_in[8];
    const float* b2  = (const float*)d_in[9];
    const float* n1w = (const float*)d_in[10];
    const float* n1b = (const float*)d_in[11];
    const float* n2w = (const float*)d_in[12];
    const float* n2b = (const float*)d_in[13];
    const float* fnw = (const float*)d_in[14];
    const float* fnb = (const float*)d_in[15];
    float* out = (float*)d_out;

    // workspace layout
    const size_t MD = (size_t)MTOK * DM;
    float* X   = (float*)d_ws;                 // MD fp32
    float* TMP = X + MD;                       // MD fp32
    u16* Xb   = (u16*)(TMP + MD);              // MD bf16
    u16* Qb   = Xb + MD;
    u16* Kb   = Qb + MD;
    u16* Vb   = Kb + MD;
    u16* ATTb = Vb + MD;
    u16* FFHb = ATTb + MD;                     // MTOK*DFF bf16
    u16* WT   = FFHb + (size_t)MTOK * DFF;     // DM*DFF bf16 (weight scratch)
    u16* EMBb = WT + (size_t)DM * DFF;         // VPAD*DM bf16

    dim3 blk256(256), blk128(128);
    dim3 gD(DM / 128, MTOK / 128);
    dim3 gF(DFF / 128, MTOK / 128);
    dim3 gV((VOCAB + 127) / 128, MTOK / 128);
    dim3 gAttn(BATCH * NH * (SEQ / 64));
    dim3 gTdd(DM / 64, DM / 64);               // cvtT for DxD
    dim3 gT1(DFF / 64, DM / 64);               // w1: K=DM, N=DFF
    dim3 gT2(DM / 64, DFF / 64);               // w2: K=DFF, N=DM

    long nEmb = (long)VOCAB * DM;
    cvt_kernel<<<(unsigned)((nEmb + 2047) / 2048), blk256, 0, stream>>>(emb, EMBb, nEmb);

    embed_kernel<<<MTOK, blk256, 0, stream>>>(ids, emb, X);

    for (int l = 0; l < NL; ++l) {
        const float* wq_l = wq + (size_t)l * DM * DM;
        const float* wk_l = wk + (size_t)l * DM * DM;
        const float* wv_l = wv + (size_t)l * DM * DM;
        const float* wo_l = wo + (size_t)l * DM * DM;
        const float* w1_l = w1 + (size_t)l * DM * DFF;
        const float* b1_l = b1 + (size_t)l * DFF;
        const float* w2_l = w2 + (size_t)l * DFF * DM;
        const float* b2_l = b2 + (size_t)l * DM;

        cvt_kernel<<<(unsigned)(MD / 2048), blk256, 0, stream>>>(X, Xb, (long)MD);

        cvtT_kernel<<<gTdd, blk256, 0, stream>>>(wq_l, WT, DM, DM);
        gemm_kernel<0, true><<<gD, blk256, 0, stream>>>(Xb, WT, nullptr, Qb, DM, DM);
        cvtT_kernel<<<gTdd, blk256, 0, stream>>>(wk_l, WT, DM, DM);
        gemm_kernel<0, true><<<gD, blk256, 0, stream>>>(Xb, WT, nullptr, Kb, DM, DM);
        cvtT_kernel<<<gTdd, blk256, 0, stream>>>(wv_l, WT, DM, DM);
        gemm_kernel<0, true><<<gD, blk256, 0, stream>>>(Xb, WT, nullptr, Vb, DM, DM);

        attn_kernel<<<gAttn, blk128, 0, stream>>>(Qb, Kb, Vb, ATTb);

        cvtT_kernel<<<gTdd, blk256, 0, stream>>>(wo_l, WT, DM, DM);
        gemm_kernel<0, false><<<gD, blk256, 0, stream>>>(ATTb, WT, nullptr, TMP, DM, DM);
        addln_kernel<true><<<MTOK, blk256, 0, stream>>>(X, TMP, n1w + (size_t)l * DM,
                                                        n1b + (size_t)l * DM);

        cvt_kernel<<<(unsigned)(MD / 2048), blk256, 0, stream>>>(X, Xb, (long)MD);
        cvtT_kernel<<<gT1, blk256, 0, stream>>>(w1_l, WT, DM, DFF);
        gemm_kernel<2, true><<<gF, blk256, 0, stream>>>(Xb, WT, b1_l, FFHb, DFF, DM);
        cvtT_kernel<<<gT2, blk256, 0, stream>>>(w2_l, WT, DFF, DM);
        gemm_kernel<1, false><<<gD, blk256, 0, stream>>>(FFHb, WT, b2_l, TMP, DM, DFF);
        addln_kernel<true><<<MTOK, blk256, 0, stream>>>(X, TMP, n2w + (size_t)l * DM,
                                                        n2b + (size_t)l * DM);
    }

    addln_kernel<false><<<MTOK, blk256, 0, stream>>>(X, nullptr, fnw, fnb);
    cvt_kernel<<<(unsigned)(MD / 2048), blk256, 0, stream>>>(X, Xb, (long)MD);

    // logits = Xb * EMBb^T  (EMBb already [V x D] = [N x K])
    gemm_kernel<0, false><<<gV, blk256, 0, stream>>>(Xb, EMBb, nullptr, out, VOCAB, DM);
}
